// Transformer_90426241450054
// MI455X (gfx1250) — matrix operand order repaired and checked by reading, not compiled, not run on hardware
//
#include <hip/hip_runtime.h>
#include <hip/hip_bf16.h>
#include <math.h>

// ---------------- problem constants ----------------
#define KVOC   512
#define ECH    256
#define HID    512
#define NHEAD  8
#define DHEAD  32
#define NLAY   2
#define NB     2       // N
#define S1C    8
#define S2C    9
#define CC     3
#define SSEQ   16      // S1 + S2 - 1
#define VOX    512     // 8*8*8
#define BSEQ   (NB*CC)                 // 6 independent sequences
#define MTOK   (BSEQ*SSEQ*VOX)         // 49152 tokens
#define CLSROWS (NB*S2C*CC*VOX)        // 27648 classifier rows
#define SCORE_PER_N (S2C*CC*VOX)       // 13824

typedef __attribute__((ext_vector_type(16))) _Float16 v16h;
typedef __attribute__((ext_vector_type(8)))  _Float16 v8h;
typedef __attribute__((ext_vector_type(8)))  float    v8f;

union V16 { v16h v; v8h h[2]; };

// ---------------- WMMA helpers ----------------
__device__ __forceinline__ v8f wmma_16x16x32(v16h a, v16h b, v8f c) {
  return __builtin_amdgcn_wmma_f32_16x16x32_f16(
      /*neg_a=*/false, a, /*neg_b=*/false, b,
      /*c_mod=*/(short)0, c, /*reuse_a=*/false, /*reuse_b=*/false);
}

// A fragment (16x32, f16): lane<16 -> row, K {0..7, 16..23}; lane>=16 -> K {8..15, 24..31}
// Unconditional: out-of-volume taps point at a dedicated zero row.
__device__ __forceinline__ v16h load_a_frag(const _Float16* __restrict__ row, int K0, int halfsel) {
  V16 u;
  u.h[0] = *(const v8h*)(row + K0 + halfsel * 8);
  u.h[1] = *(const v8h*)(row + K0 + 16 + halfsel * 8);
  return u.v;
}

// B fragment (32x16, f16) from K-contiguous column storage: lane holds column,
// halves are K = K0 + halfsel*8 + {0..7} and K0 + 16 + halfsel*8 + {0..7}.
__device__ __forceinline__ v16h load_b_frag(const _Float16* __restrict__ col, int K0, int halfsel) {
  V16 u;
  u.h[0] = *(const v8h*)(col + K0 + halfsel * 8);
  u.h[1] = *(const v8h*)(col + K0 + 16 + halfsel * 8);
  return u.v;
}

__device__ __forceinline__ float gelu_exact(float x) {
  return 0.5f * x * (1.0f + erff(x * 0.70710678118654752f));
}

// ---------------- table builders ----------------
// Out-of-volume taps -> zero-row index MTOK (buffers are allocated with MTOK+1 rows).
__global__ void k_build_nbr(int* __restrict__ nbr) {
  int idx = blockIdx.x * 256 + threadIdx.x;
  if (idx >= MTOK * 27) return;
  int m = idx / 27, t = idx % 27;
  int vox = m & (VOX - 1);
  int bs  = m >> 9;                 // b*16 + s
  int w = vox & 7, v = (vox >> 3) & 7, u = vox >> 6;
  int dw = t % 3 - 1, dv = (t / 3) % 3 - 1, du = t / 9 - 1;
  int uu = u + du, vv = v + dv, ww = w + dw;
  nbr[idx] = (uu >= 0 && uu < 8 && vv >= 0 && vv < 8 && ww >= 0 && ww < 8)
                 ? (bs * VOX + (uu << 6) + (vv << 3) + ww) : MTOK;
}

__global__ void k_build_cls(int* __restrict__ clsrow, int* __restrict__ idrow) {
  int r = blockIdx.x * 256 + threadIdx.x;
  if (r >= CLSROWS) return;
  int vox  = r & (VOX - 1);
  int c    = (r >> 9) % 3;
  int srel = (r / (VOX * CC)) % S2C;
  int n    = r / SCORE_PER_N;
  clsrow[r] = (((n * CC + c) * SSEQ) + (S1C - 1) + srel) * VOX + vox;  // s = 7+srel
  idrow[r]  = r;
}

// Zero the padding rows of the conv-input activation buffers.
__global__ void k_zero_pad(_Float16* __restrict__ hf16, _Float16* __restrict__ of16,
                           _Float16* __restrict__ g16) {
  int i = blockIdx.x * 256 + threadIdx.x;   // 1024 threads total
  if (i < 256)       hf16[(size_t)MTOK * ECH + i]       = (_Float16)0.0f;
  else if (i < 512)  of16[(size_t)MTOK * ECH + (i-256)] = (_Float16)0.0f;
  else if (i < 1024) g16 [(size_t)MTOK * HID + (i-512)] = (_Float16)0.0f;
}

// ---------------- weight convert + transpose: src [T][CI][CO] f32 -> dst [T][CO][CI] f16 ----------------
__global__ void k_cvt_w(const float* __restrict__ src, _Float16* __restrict__ dst,
                        int T, int CI, int CO) {
  long idx = (long)blockIdx.x * 256 + threadIdx.x;
  long tot = (long)T * CI * CO;
  if (idx >= tot) return;
  int  ci = (int)(idx % CI);
  long r  = idx / CI;
  int  co = (int)(r % CO);
  int  t  = (int)(r / CO);
  dst[idx] = (_Float16)src[((long)t * CI + ci) * CO + co];
}

// ---------------- embedding ----------------
__global__ void k_embed(const int* __restrict__ code, const int* __restrict__ ncode,
                        const float* __restrict__ emb, const float* __restrict__ pos,
                        float* __restrict__ hf32, _Float16* __restrict__ hf16) {
  long idx = (long)blockIdx.x * 256 + threadIdx.x;
  if (idx >= (long)MTOK * ECH) return;
  int e = (int)(idx & (ECH - 1));
  int m = (int)(idx >> 8);
  int vox = m & (VOX - 1);
  int bs  = m >> 9;
  int s = bs & (SSEQ - 1);
  int b = bs >> 4;
  int c = b % CC, n = b / CC;
  int id;
  if (s < S1C) id = code [(((n * S1C + s)        * CC + c) << 9) + vox];
  else         id = ncode[(((n * S2C + (s - S1C)) * CC + c) << 9) + vox];
  float h = emb[(long)id * ECH + e] + pos[s * ECH + e];
  hf32[idx] = h;
  hf16[idx] = (_Float16)h;
}

// ---------------- generic tapped GEMM (implicit conv / dense) ----------------
// out[m, co] = sum_t sum_ci act[nbr[m,t], ci] * wt[t, co, ci]  (+ bias, optional gelu)
// Block = 8 waves sharing one 64-wide N range (B reuse through WGP$), each wave
// owns a 32x64 tile: 2 A frags + 4 B frags -> 8 independent WMMA accumulators
// per K-chunk (halves B traffic vs 16x64, hides the full XDL pipeline depth).
template<int CIN>
__global__ void k_gemm_tap(const _Float16* __restrict__ act, const _Float16* __restrict__ wt,
                           const float* __restrict__ bias, const int* __restrict__ nbr,
                           int T, int COUT, int Mrows,
                           float* __restrict__ out32, _Float16* __restrict__ out16,
                           int do_gelu, int score_map) {
  int wid  = threadIdx.x >> 5;
  int lane = threadIdx.x & 31;
  int halfsel = lane >> 4;
  int lx   = lane & 15;
  int ntiles = COUT >> 6;                 // tiles of 64 output channels
  int mg = blockIdx.x / ntiles;           // group of 256 rows (8 waves x 32)
  int nt = blockIdx.x % ntiles;
  int M0 = (mg * 8 + wid) << 5;           // 32 rows per wave
  int N0 = nt << 6;

  const int* nrow0 = nbr + (size_t)(M0 + lx) * T;
  const int* nrow1 = nbr + (size_t)(M0 + 16 + lx) * T;
  const _Float16* bcol0 = wt + (size_t)(N0 + lx) * CIN;
  const size_t tapstride = (size_t)COUT * CIN;
  const size_t colstep   = (size_t)16 * CIN;

  v8f acc[2][4];
#pragma unroll
  for (int i = 0; i < 2; ++i)
#pragma unroll
    for (int j = 0; j < 4; ++j) acc[i][j] = (v8f){};

  for (int t = 0; t < T; ++t) {
    const _Float16* arow0 = act + (size_t)nrow0[t] * CIN;
    const _Float16* arow1 = act + (size_t)nrow1[t] * CIN;
    const _Float16* bcol  = bcol0 + (size_t)t * tapstride;
    if (t + 1 < T)
      __builtin_prefetch(bcol0 + (size_t)(t + 1) * tapstride, 0, 1);
#pragma unroll 4
    for (int K0 = 0; K0 < CIN; K0 += 32) {
      v16h a0 = load_a_frag(arow0, K0, halfsel);
      v16h a1 = load_a_frag(arow1, K0, halfsel);
      v16h b0 = load_b_frag(bcol,               K0, halfsel);
      v16h b1 = load_b_frag(bcol +     colstep, K0, halfsel);
      v16h b2 = load_b_frag(bcol + 2 * colstep, K0, halfsel);
      v16h b3 = load_b_frag(bcol + 3 * colstep, K0, halfsel);
      acc[0][0] = wmma_16x16x32(a0, b0, acc[0][0]);
      acc[1][0] = wmma_16x16x32(a1, b0, acc[1][0]);
      acc[0][1] = wmma_16x16x32(a0, b1, acc[0][1]);
      acc[1][1] = wmma_16x16x32(a1, b1, acc[1][1]);
      acc[0][2] = wmma_16x16x32(a0, b2, acc[0][2]);
      acc[1][2] = wmma_16x16x32(a1, b2, acc[1][2]);
      acc[0][3] = wmma_16x16x32(a0, b3, acc[0][3]);
      acc[1][3] = wmma_16x16x32(a1, b3, acc[1][3]);
    }
  }

#pragma unroll
  for (int j = 0; j < 4; ++j) {
    int n = N0 + j * 16 + lx;
    float bval = bias ? bias[n] : 0.0f;
#pragma unroll
    for (int i = 0; i < 2; ++i) {
#pragma unroll
      for (int r = 0; r < 8; ++r) {
        int m = M0 + i * 16 + r + (halfsel << 3);
        float x = acc[i][j][r] + bval;
        if (do_gelu) x = gelu_exact(x);
        if (out32) {
          size_t o;
          if (score_map) {
            // m = ((n_*9+srel)*3+c)*512+vox -> score[((n_*K + n)*9+srel)*1536 + c*512+vox]
            int nn   = m / SCORE_PER_N;
            int rem  = m % SCORE_PER_N;
            int srel = rem / (VOX * CC);
            int rem2 = rem % (VOX * CC);
            o = ((size_t)nn * KVOC + n) * SCORE_PER_N + (size_t)srel * (VOX * CC) + rem2;
          } else {
            o = (size_t)m * COUT + n;
          }
          out32[o] = x;
        }
        if (out16) out16[(size_t)m * COUT + n] = (_Float16)x;
      }
    }
  }
  (void)Mrows;
}

// ---------------- attention: one wave per (b, head, voxel) ----------------
__global__ void k_attn(const _Float16* __restrict__ q, const _Float16* __restrict__ k,
                       const _Float16* __restrict__ v, _Float16* __restrict__ o) {
  __shared__ __align__(16) _Float16 lds[8][SSEQ * 32];
  int wid  = threadIdx.x >> 5;
  int lane = threadIdx.x & 31;
  int halfsel = lane >> 4;
  int lx   = lane & 15;
  int wg = blockIdx.x * 8 + wid;          // BSEQ*NHEAD*VOX = 24576 waves, exact grid
  int vox = wg & (VOX - 1);
  int bh  = wg >> 9;
  int hh  = bh & (NHEAD - 1);
  int b   = bh >> 3;
  size_t tokbase = ((size_t)b * SSEQ) * VOX + vox;   // token(s) = tokbase + s*VOX
  size_t chan    = (size_t)hh * DHEAD;
  const size_t tstride = (size_t)VOX * ECH;          // halves between seq steps

  // Q as A-frag (16 x 32): row s = lx, K = head dim (contiguous)
  const _Float16* qrow = q + (tokbase + (size_t)lx * VOX) * ECH + chan;
  V16 ua;
  ua.h[0] = *(const v8h*)(qrow + halfsel * 8);
  ua.h[1] = *(const v8h*)(qrow + 16 + halfsel * 8);
  // K^T as B-frag (32 x 16): column t = lx, K = head dim (contiguous)
  const _Float16* kcol = k + (tokbase + (size_t)lx * VOX) * ECH + chan;
  V16 ub;
  ub.h[0] = *(const v8h*)(kcol + halfsel * 8);
  ub.h[1] = *(const v8h*)(kcol + 16 + halfsel * 8);

  v8f sc = {};
  sc = wmma_16x16x32(ua.v, ub.v, sc);

  const float scale = 0.17677669529663687f;  // 1/sqrt(32)
  float pr[8];
#pragma unroll
  for (int r = 0; r < 8; ++r) {
    int srow = r + (halfsel << 3);
    int tcol = lx;
    float x = sc[r] * scale;
    bool vis = (tcol < S1C) || (tcol <= srow);
    x = vis ? x : -1e30f;
    float mx = x;
#pragma unroll
    for (int d = 1; d < 16; d <<= 1) mx = fmaxf(mx, __shfl_xor(mx, d, 32));
    float ex = expf(x - mx);
    float sm = ex;
#pragma unroll
    for (int d = 1; d < 16; d <<= 1) sm += __shfl_xor(sm, d, 32);
    pr[r] = ex / sm;
  }

  // Re-stripe probabilities into A-frag layout via LDS (K padded 16 -> 32 with zeros)
  _Float16* L = lds[wid];
#pragma unroll
  for (int r = 0; r < 8; ++r) {
    int srow = r + (halfsel << 3);
    L[srow * 32 + lx]      = (_Float16)pr[r];
    L[srow * 32 + 16 + lx] = (_Float16)0.0f;
  }
  asm volatile("s_wait_dscnt 0" ::: "memory");

  V16 pa;
  pa.h[0] = *(const v8h*)(L + lx * 32 + halfsel * 16);
  pa.h[1] = *(const v8h*)(L + lx * 32 + halfsel * 16 + 8);

  // o = attn(16x32 padded) * V ; V as B-frag with strided (seq) K, two 16-col halves
#pragma unroll
  for (int nsub = 0; nsub < 2; ++nsub) {
    const _Float16* vbase = v + tokbase * ECH + chan + nsub * 16 + lx;
    V16 vb;
#pragma unroll
    for (int j = 0; j < 16; ++j) {
      int t = halfsel * 16 + j;
      vb.v[j] = (t < SSEQ) ? vbase[(size_t)t * tstride] : (_Float16)0.0f;
    }
    v8f oacc = {};
    oacc = wmma_16x16x32(pa.v, vb.v, oacc);
#pragma unroll
    for (int r = 0; r < 8; ++r) {
      int s = r + (halfsel << 3);
      o[(tokbase + (size_t)s * VOX) * ECH + chan + nsub * 16 + lx] = (_Float16)oacc[r];
    }
  }
}

// ---------------- (residual) + LayerNorm (+gelu) : one wave per row of 256 ----------------
__global__ void k_ln(const float* __restrict__ x, const float* __restrict__ res,
                     const float* __restrict__ g, const float* __restrict__ b,
                     float* __restrict__ out32, _Float16* __restrict__ out16,
                     int Mrows, int gelu_after) {
  int wid = threadIdx.x >> 5, lane = threadIdx.x & 31;
  int row = blockIdx.x * 8 + wid;
  if (row >= Mrows) return;
  const float* xp = x + (size_t)row * ECH;
  const float* rp = res ? res + (size_t)row * ECH : (const float*)0;
  float vals[8];
  float s = 0.0f;
#pragma unroll
  for (int i = 0; i < 8; ++i) {
    int c = lane + 32 * i;
    float vv = xp[c] + (rp ? rp[c] : 0.0f);
    vals[i] = vv; s += vv;
  }
#pragma unroll
  for (int d = 16; d >= 1; d >>= 1) s += __shfl_xor(s, d, 32);
  float mu = s * (1.0f / ECH);
  float var = 0.0f;
#pragma unroll
  for (int i = 0; i < 8; ++i) { float dd = vals[i] - mu; var += dd * dd; }
#pragma unroll
  for (int d = 16; d >= 1; d >>= 1) var += __shfl_xor(var, d, 32);
  var *= (1.0f / ECH);
  float inv = rsqrtf(var + 1e-5f);
#pragma unroll
  for (int i = 0; i < 8; ++i) {
    int c = lane + 32 * i;
    float y = (vals[i] - mu) * inv * g[c] + b[c];
    if (gelu_after) y = gelu_exact(y);
    if (out32) out32[(size_t)row * ECH + c] = y;
    if (out16) out16[(size_t)row * ECH + c] = (_Float16)y;
  }
}

// ---------------- loss / pred ----------------
__global__ void k_zero_loss(float* __restrict__ loss) {
  if (blockIdx.x == 0 && threadIdx.x == 0) *loss = 0.0f;
}

__global__ void k_loss(const float* __restrict__ score, const int* __restrict__ ncode,
                       float* __restrict__ loss, float* __restrict__ pred) {
  int r = blockIdx.x * 256 + threadIdx.x;
  if (r >= CLSROWS) return;
  int rem2 = r % (VOX * CC);          // c*512+vox
  int srel = (r / (VOX * CC)) % S2C;
  int n    = r / SCORE_PER_N;
  const float* sp = score + (size_t)n * KVOC * SCORE_PER_N
                          + (size_t)srel * (VOX * CC) + rem2;
  float mx = -1e30f; int am = 0;
  for (int kk = 0; kk < KVOC; ++kk) {
    float x = sp[(size_t)kk * SCORE_PER_N];
    if (x > mx) { mx = x; am = kk; }
  }
  float se = 0.0f;
  for (int kk = 0; kk < KVOC; ++kk)
    se += expf(sp[(size_t)kk * SCORE_PER_N] - mx);
  float lse = mx + logf(se);
  int lbl = ncode[r];  // ncode flat index == r by construction
  float nll = lse - sp[(size_t)lbl * SCORE_PER_N];
  atomicAdd(loss, nll * (1.0f / CLSROWS));
  pred[r] = (float)am;
}

// ---------------- host side ----------------
static inline char* carve(char*& p, size_t bytes) {
  char* r = p;
  p += (bytes + 255) & ~(size_t)255;
  return r;
}
static inline int gemm_grid(int Mrows, int COUT) {
  return (Mrows / 256) * (COUT / 64);   // Mrows % 256 == 0 for all uses
}

extern "C" void kernel_launch(void* const* d_in, const int* in_sizes, int n_in,
                              void* d_out, int out_size, void* d_ws, size_t ws_size,
                              hipStream_t stream) {
  (void)in_sizes; (void)n_in; (void)out_size; (void)ws_size;
  const int*   code    = (const int*)  d_in[0];
  const int*   ncode   = (const int*)  d_in[1];
  const float* emb     = (const float*)d_in[2];
  const float* pos_emb = (const float*)d_in[3];
  const float* Wq = (const float*)d_in[4];   const float* bq = (const float*)d_in[5];
  const float* Wk = (const float*)d_in[6];   const float* bk = (const float*)d_in[7];
  const float* Wv = (const float*)d_in[8];   const float* bv = (const float*)d_in[9];
  const float* Wo = (const float*)d_in[10];  const float* bo = (const float*)d_in[11];
  const float* ln1_g = (const float*)d_in[12]; const float* ln1_b = (const float*)d_in[13];
  const float* ln2_g = (const float*)d_in[14]; const float* ln2_b = (const float*)d_in[15];
  const float* Wc1 = (const float*)d_in[16]; const float* bc1 = (const float*)d_in[17];
  const float* Wc2 = (const float*)d_in[18]; const float* bc2 = (const float*)d_in[19];
  const float* cls_W1 = (const float*)d_in[20]; const float* cls_b1 = (const float*)d_in[21];
  const float* cls_ln_g = (const float*)d_in[22]; const float* cls_ln_b = (const float*)d_in[23];
  const float* cls_W2 = (const float*)d_in[24]; const float* cls_b2 = (const float*)d_in[25];

  float* score = (float*)d_out;                          // 2*512*9*3*512 floats
  float* loss  = score + (size_t)NB * KVOC * SCORE_PER_N;
  float* pred  = loss + 1;

  // workspace carve (conv-input f16 buffers get one extra zero row)
  char* p = (char*)d_ws;
  size_t rowsE  = (size_t)MTOK * ECH;
  size_t rowsEp = (size_t)(MTOK + 1) * ECH;
  float*     hf32 = (float*)    carve(p, rowsE * 4);
  float*     t32  = (float*)    carve(p, rowsE * 4);
  _Float16*  hf16 = (_Float16*) carve(p, rowsEp * 2);
  _Float16*  qf16 = (_Float16*) carve(p, rowsE * 2);
  _Float16*  kf16 = (_Float16*) carve(p, rowsE * 2);
  _Float16*  vf16 = (_Float16*) carve(p, rowsE * 2);
  _Float16*  of16 = (_Float16*) carve(p, rowsEp * 2);
  _Float16*  g16  = (_Float16*) carve(p, (size_t)(MTOK + 1) * HID * 2);
  const size_t WEE = 27u * ECH * ECH;          // 1,769,472 halves
  const size_t WEH = 27u * ECH * HID;          // 3,538,944 halves
  const size_t LW  = 4 * WEE + 2 * WEH;        // per layer
  _Float16*  wbuf = (_Float16*) carve(p, (2 * LW + (size_t)ECH * ECH + (size_t)ECH * KVOC) * 2);
  int* nbr    = (int*) carve(p, (size_t)MTOK * 27 * 4);
  int* clsrow = (int*) carve(p, (size_t)CLSROWS * 4);
  int* idrow  = (int*) carve(p, (size_t)CLSROWS * 4);

  // -------- prep: tables, zero pad rows, weight convert, embedding --------
  k_build_nbr<<<(MTOK * 27 + 255) / 256, 256, 0, stream>>>(nbr);
  k_build_cls<<<(CLSROWS + 255) / 256, 256, 0, stream>>>(clsrow, idrow);
  k_zero_pad<<<4, 256, 0, stream>>>(hf16, of16, g16);

  _Float16* wq16[NLAY]; _Float16* wk16[NLAY]; _Float16* wv16[NLAY];
  _Float16* wo16[NLAY]; _Float16* wc116[NLAY]; _Float16* wc216[NLAY];
  for (int l = 0; l < NLAY; ++l) {
    _Float16* base = wbuf + (size_t)l * LW;
    wq16[l]  = base;
    wk16[l]  = base + WEE;
    wv16[l]  = base + 2 * WEE;
    wo16[l]  = base + 3 * WEE;
    wc116[l] = base + 4 * WEE;
    wc216[l] = base + 4 * WEE + WEH;
    int blkEE = (int)((WEE + 255) / 256);
    int blkEH = (int)((WEH + 255) / 256);
    k_cvt_w<<<blkEE, 256, 0, stream>>>(Wq + (size_t)l * WEE, wq16[l], 27, ECH, ECH);
    k_cvt_w<<<blkEE, 256, 0, stream>>>(Wk + (size_t)l * WEE, wk16[l], 27, ECH, ECH);
    k_cvt_w<<<blkEE, 256, 0, stream>>>(Wv + (size_t)l * WEE, wv16[l], 27, ECH, ECH);
    k_cvt_w<<<blkEE, 256, 0, stream>>>(Wo + (size_t)l * WEE, wo16[l], 27, ECH, ECH);
    k_cvt_w<<<blkEH, 256, 0, stream>>>(Wc1 + (size_t)l * WEH, wc116[l], 27, ECH, HID);
    k_cvt_w<<<blkEH, 256, 0, stream>>>(Wc2 + (size_t)l * WEH, wc216[l], 27, HID, ECH);
  }
  _Float16* w1c = wbuf + 2 * LW;
  _Float16* w2c = w1c + (size_t)ECH * ECH;
  k_cvt_w<<<(ECH * ECH + 255) / 256, 256, 0, stream>>>(cls_W1, w1c, 1, ECH, ECH);
  k_cvt_w<<<(ECH * KVOC + 255) / 256, 256, 0, stream>>>(cls_W2, w2c, 1, ECH, KVOC);

  k_embed<<<(int)((rowsE + 255) / 256), 256, 0, stream>>>(code, ncode, emb, pos_emb, hf32, hf16);

  // -------- transformer layers --------
  for (int l = 0; l < NLAY; ++l) {
    // Q/K/V convs (f16 outputs only)
    k_gemm_tap<ECH><<<gemm_grid(MTOK, ECH), 256, 0, stream>>>(
        hf16, wq16[l], bq + l * ECH, nbr, 27, ECH, MTOK, (float*)0, qf16, 0, 0);
    k_gemm_tap<ECH><<<gemm_grid(MTOK, ECH), 256, 0, stream>>>(
        hf16, wk16[l], bk + l * ECH, nbr, 27, ECH, MTOK, (float*)0, kf16, 0, 0);
    k_gemm_tap<ECH><<<gemm_grid(MTOK, ECH), 256, 0, stream>>>(
        hf16, wv16[l], bv + l * ECH, nbr, 27, ECH, MTOK, (float*)0, vf16, 0, 0);
    // attention: 24576 waves / 8 = 3072 blocks (exact)
    k_attn<<<BSEQ * NHEAD * VOX / 8, 256, 0, stream>>>(qf16, kf16, vf16, of16);
    // output projection conv (fp32) then residual + LN1
    k_gemm_tap<ECH><<<gemm_grid(MTOK, ECH), 256, 0, stream>>>(
        of16, wo16[l], bo + l * ECH, nbr, 27, ECH, MTOK, t32, (_Float16*)0, 0, 0);
    k_ln<<<(MTOK + 7) / 8, 256, 0, stream>>>(
        t32, hf32, ln1_g + l * ECH, ln1_b + l * ECH, hf32, hf16, MTOK, 0);
    // FFN: conv1 (gelu, f16) -> conv2 (fp32) -> residual + LN2
    k_gemm_tap<ECH><<<gemm_grid(MTOK, HID), 256, 0, stream>>>(
        hf16, wc116[l], bc1 + l * HID, nbr, 27, HID, MTOK, (float*)0, g16, 1, 0);
    k_gemm_tap<HID><<<gemm_grid(MTOK, ECH), 256, 0, stream>>>(
        g16, wc216[l], bc2 + l * ECH, nbr, 27, ECH, MTOK, t32, (_Float16*)0, 0, 0);
    k_ln<<<(MTOK + 7) / 8, 256, 0, stream>>>(
        t32, hf32, ln2_g + l * ECH, ln2_b + l * ECH, hf32, hf16, MTOK, 0);
  }

  // -------- classifier --------
  // z1 = h_last @ W1 + b1  (gather rows via clsrow, dense T=1)
  k_gemm_tap<ECH><<<gemm_grid(CLSROWS, ECH), 256, 0, stream>>>(
      hf16, w1c, cls_b1, clsrow, 1, ECH, CLSROWS, t32, (_Float16*)0, 0, 0);
  // gelu(ln(z1)) -> f16
  k_ln<<<(CLSROWS + 7) / 8, 256, 0, stream>>>(
      t32, (const float*)0, cls_ln_g, cls_ln_b, (float*)0, g16, CLSROWS, 1);
  // score = z @ W2 + b2, stored with (n, K, s2, c, u, v, w) permutation
  k_gemm_tap<ECH><<<gemm_grid(CLSROWS, KVOC), 256, 0, stream>>>(
      g16, w2c, cls_b2, idrow, 1, KVOC, CLSROWS, score, (_Float16*)0, 0, 1);

  // -------- loss & pred --------
  k_zero_loss<<<1, 64, 0, stream>>>(loss);
  k_loss<<<(CLSROWS + 255) / 256, 256, 0, stream>>>(score, ncode, loss, pred);
}
